// fus_moe_add_44942537786084
// MI455X (gfx1250) — matrix operand order repaired and checked by reading, not compiled, not run on hardware
//
#include <hip/hip_runtime.h>

typedef _Float16 f16;
typedef __attribute__((ext_vector_type(16))) _Float16 v16h;
typedef __attribute__((ext_vector_type(8)))  float    v8f;

#define Bsz 8192
#define Dd  256
#define Tt  3

// ---------------------------------------------------------------------------
// Weight convert: src [nmat][K][N] f32 -> dst [nmat][N][K] f16 (transposed)
// ---------------------------------------------------------------------------
__global__ void convw_kernel(const float* __restrict__ src, f16* __restrict__ dst,
                             int K, int N, long total) {
  long j = (long)blockIdx.x * blockDim.x + threadIdx.x;
  if (j >= total) return;
  long kn = (long)K * N;
  long m  = j / kn;
  long r  = j - m * kn;
  int  n  = (int)(r / K);
  int  k  = (int)(r - (long)n * K);
  dst[j] = (f16)src[m * kn + (long)k * N + n];
}

// ---------------------------------------------------------------------------
// Embedding gather: X[t][b][f*16+j] = emb[f][x[b,f]][j] for t = 0..2
// ---------------------------------------------------------------------------
__global__ void embed_kernel(const int* __restrict__ x, const float* __restrict__ emb,
                             f16* __restrict__ X) {
  int i = blockIdx.x * blockDim.x + threadIdx.x;   // over B*F
  if (i >= Bsz * 16) return;
  int b = i >> 4, f = i & 15;
  int v = x[i];
  const float* s = emb + ((long)f * 10000 + v) * 16;
  long base = (long)b * Dd + f * 16;
  for (int j = 0; j < 16; ++j) {
    f16 h = (f16)s[j];
    X[base + j]                 = h;
    X[base + (long)Bsz*Dd + j]  = h;
    X[base + 2L*Bsz*Dd + j]     = h;
  }
}

// ---------------------------------------------------------------------------
// WMMA GEMM: C[M x N] = act(A[M x K] @ Wt^T + bias), Wt stored [N x K] f16.
// Block = 256 threads = 8 waves; wave w owns rows [blockM + w*32, +32),
// cols [blockN, +64): 2 A fragments x 4 B fragments = 8 WMMAs per k-step.
// W tile (32K x 64N) staged in LDS; next tile's global load is issued before
// the compute phase so its latency hides behind the 8 WMMAs (software pipe).
// Requires M%256==0, N%64==0, K%32==0. EXEC all-ones everywhere (WMMA req).
// ---------------------------------------------------------------------------
#define LDS_STRIDE 40   // halves per row; 80B pitch: 16B aligned, conflict-free
__launch_bounds__(256)
__global__ void gemm_wmma(const f16* __restrict__ A, const f16* __restrict__ Wt,
                          const float* __restrict__ bias, f16* __restrict__ outH,
                          float* __restrict__ outF, int M, int N, int K, int relu) {
  __shared__ f16 lb[64 * LDS_STRIDE];
  const int tid  = threadIdx.x;
  const int wave = tid >> 5;
  const int lane = tid & 31;
  const int l16  = lane & 15;
  const int hi   = lane >> 4;            // 0: lanes 0-15, 1: lanes 16-31
  const int n0   = blockIdx.x * 64;
  const int m0   = blockIdx.y * 256 + wave * 32;

  const f16* arow0 = A + (long)(m0 + l16) * K;        // A rows owned by lane
  const f16* arow1 = A + (long)(m0 + 16 + l16) * K;

  // cooperative W-tile load mapping: 256 threads x 8 halves = 64x32 tile
  const int wn = tid >> 2;               // 0..63  (local n)
  const int wk = (tid & 3) * 8;          // 0,8,16,24 (local k)
  const f16* wsrc = Wt + (long)(n0 + wn) * K + wk;

  v8f acc[2][4];
  for (int g = 0; g < 2; ++g)
    for (int a = 0; a < 4; ++a)
      acc[g][a] = (v8f){0.f,0.f,0.f,0.f,0.f,0.f,0.f,0.f};

  const int nk = K / 32;
  uint4 wreg = *(const uint4*)(wsrc);                 // stage tile 0 in regs

  for (int ki = 0; ki < nk; ++ki) {
    __syncthreads();                                  // readers of lb are done
    *(uint4*)&lb[wn * LDS_STRIDE + wk] = wreg;        // 16B-aligned ds_store
    __syncthreads();                                  // tile visible

    // issue NEXT W tile load now; its s_wait lands at next iter's ds_store
    if (ki + 1 < nk) wreg = *(const uint4*)(wsrc + (ki + 1) * 32);
    if (ki + 2 < nk) __builtin_prefetch(wsrc + (ki + 2) * 32, 0, 1);

    // A fragments: lane holds K = kb..kb+7 and kb+16..kb+23 of its rows
    const int kb = ki * 32 + hi * 8;
    union { uint4 u[2]; v16h h; } a0f, a1f;
    a0f.u[0] = *(const uint4*)(arow0 + kb);
    a0f.u[1] = *(const uint4*)(arow0 + kb + 16);
    a1f.u[0] = *(const uint4*)(arow1 + kb);
    a1f.u[1] = *(const uint4*)(arow1 + kb + 16);
    if (ki + 1 < nk) {
      __builtin_prefetch(arow0 + kb + 32, 0, 1);
      __builtin_prefetch(arow1 + kb + 32, 0, 1);
    }

    for (int a = 0; a < 4; ++a) {
      // B fragment: lane holds column a*16+l16, local K hi*8..+7 and 16+hi*8..+7
      const f16* bp = &lb[(a * 16 + l16) * LDS_STRIDE + hi * 8];
      union { uint2 u[4]; v16h h; } bf_;
      bf_.u[0] = *(const uint2*)(bp);
      bf_.u[1] = *(const uint2*)(bp + 4);
      bf_.u[2] = *(const uint2*)(bp + 16);
      bf_.u[3] = *(const uint2*)(bp + 20);
      acc[0][a] = __builtin_amdgcn_wmma_f32_16x16x32_f16(
          false, a0f.h, false, bf_.h, (short)0, acc[0][a], false, false);
      acc[1][a] = __builtin_amdgcn_wmma_f32_16x16x32_f16(
          false, a1f.h, false, bf_.h, (short)0, acc[1][a], false, false);
    }
  }

  // Epilogue: VGPR i -> M = m0 + g*16 + i + hi*8, N = n0 + a*16 + l16
  for (int a = 0; a < 4; ++a) {
    int col  = n0 + a * 16 + l16;
    float bv = bias ? bias[col] : 0.f;
    for (int g = 0; g < 2; ++g) {
      for (int i = 0; i < 8; ++i) {
        int r   = m0 + g * 16 + i + hi * 8;
        float v = acc[g][a][i] + bv;
        if (relu) v = fmaxf(v, 0.f);
        if (outF) outF[(long)r * N + col] = v;
        if (outH) outH[(long)r * N + col] = (f16)v;
      }
    }
  }
}

// ---------------------------------------------------------------------------
// Gate/combine: one wave per (t,b). Computes gate softmax (NE=4), cgc,
// task-gate logits -> idx2 (argmax off-diagonal), writes cgc f32 + pair[n=0].
// ---------------------------------------------------------------------------
__launch_bounds__(256)
__global__ void combine_kernel(const f16* __restrict__ X,
                               const float* __restrict__ Wg, const float* __restrict__ bg,
                               const float* __restrict__ Wf, const float* __restrict__ bf,
                               const f16* __restrict__ e2, float* __restrict__ cgc,
                               f16* __restrict__ pair, int* __restrict__ idx2) {
  int gw   = blockIdx.x * 8 + (threadIdx.x >> 5);   // over T*B
  int lane = threadIdx.x & 31;
  int t = gw / Bsz, b = gw - t * Bsz;

  const f16* xp = X + (long)gw * Dd;
  float xv[8];
  for (int j = 0; j < 8; ++j) xv[j] = (float)xp[lane * 8 + j];

  const float* wg = Wg + (long)t * Dd * 4;
  const float* wf = Wf + (long)t * Dd * 3;
  float g[4], fl[3];
  for (int e = 0; e < 4; ++e) {
    float s = 0.f;
    for (int j = 0; j < 8; ++j) s += xv[j] * wg[(lane * 8 + j) * 4 + e];
    g[e] = s;
  }
  for (int s2 = 0; s2 < 3; ++s2) {
    float s = 0.f;
    for (int j = 0; j < 8; ++j) s += xv[j] * wf[(lane * 8 + j) * 3 + s2];
    fl[s2] = s;
  }
  for (int off = 16; off; off >>= 1) {
    for (int e = 0; e < 4; ++e)  g[e]  += __shfl_xor(g[e],  off, 32);
    for (int s2 = 0; s2 < 3; ++s2) fl[s2] += __shfl_xor(fl[s2], off, 32);
  }
  for (int e = 0; e < 4; ++e)  g[e]  += bg[t * 4 + e];
  for (int s2 = 0; s2 < 3; ++s2) fl[s2] += bf[t * 3 + s2];

  // gate softmax
  float mx = fmaxf(fmaxf(g[0], g[1]), fmaxf(g[2], g[3]));
  float es = 0.f;
  for (int e = 0; e < 4; ++e) { g[e] = __expf(g[e] - mx); es += g[e]; }
  float inv = 1.f / es;
  for (int e = 0; e < 4; ++e) g[e] *= inv;

  // argmax over s != t (softmax is monotonic -> argmax of logits; first-max wins)
  int best = -1; float bv = -INFINITY;
  for (int s2 = 0; s2 < 3; ++s2) {
    if (s2 == t) continue;
    if (fl[s2] > bv) { bv = fl[s2]; best = s2; }
  }
  if (lane == 0) idx2[gw] = best;

  float* cp = cgc + (long)gw * Dd;
  f16*   pp = pair + (long)gw * 2 * Dd;   // n = 0 slot
  for (int j = 0; j < 8; ++j) {
    int h = lane + 32 * j;
    float s = 0.f;
    for (int e = 0; e < 4; ++e)
      s += g[e] * (float)e2[(((long)t * 4 + e) * Bsz + b) * Dd + h];
    cp[h] = s;
    pp[h] = (f16)s;
  }
}

// pair[t,b,1,:] = cgc[idx2[t,b], b, :]
__global__ void gather_kernel(const float* __restrict__ cgc, const int* __restrict__ idx2,
                              f16* __restrict__ pair) {
  int i = blockIdx.x * blockDim.x + threadIdx.x;   // over T*B*256
  if (i >= Tt * Bsz * Dd) return;
  int h = i & 255; int tb = i >> 8; int b = tb % Bsz;
  int s = idx2[tb];
  pair[((long)tb * 2 + 1) * Dd + h] = (f16)cgc[((long)s * Bsz + b) * Dd + h];
}

// ---------------------------------------------------------------------------
// Attention combine: one wave per (t,b). a = softmax_n(Q_n.K_n/16);
// Xout = a0*V0 + a1*V1 (f16, fed to next level).
// ---------------------------------------------------------------------------
__launch_bounds__(256)
__global__ void attn_kernel(const f16* __restrict__ Q, const f16* __restrict__ K,
                            const f16* __restrict__ V, f16* __restrict__ Xout) {
  int gw   = blockIdx.x * 8 + (threadIdx.x >> 5);   // over T*B
  int lane = threadIdx.x & 31;
  long base = (long)gw * 2 * Dd;
  float s0 = 0.f, s1 = 0.f, v0h[8], v1h[8];
  for (int j = 0; j < 8; ++j) {
    int h = lane + 32 * j;
    s0 += (float)Q[base + h]      * (float)K[base + h];
    s1 += (float)Q[base + Dd + h] * (float)K[base + Dd + h];
    v0h[j] = (float)V[base + h];
    v1h[j] = (float)V[base + Dd + h];
  }
  for (int off = 16; off; off >>= 1) {
    s0 += __shfl_xor(s0, off, 32);
    s1 += __shfl_xor(s1, off, 32);
  }
  s0 *= 0.0625f; s1 *= 0.0625f;                     // / sqrt(256)
  float m  = fmaxf(s0, s1);
  float e0 = __expf(s0 - m), e1 = __expf(s1 - m);
  float inv = 1.f / (e0 + e1);
  float a0 = e0 * inv, a1 = e1 * inv;
  for (int j = 0; j < 8; ++j) {
    int h = lane + 32 * j;
    Xout[(long)gw * Dd + h] = (f16)(a0 * v0h[j] + a1 * v1h[j]);
  }
}

// Final tower head: y = h2 @ Wt3 + bt3; sigmoid for t=0,1; out[b,3]
__global__ void tower_out_kernel(const f16* __restrict__ h2, const float* __restrict__ Wt3,
                                 const float* __restrict__ bt3, float* __restrict__ out) {
  int i = blockIdx.x * blockDim.x + threadIdx.x;    // over T*B
  if (i >= Tt * Bsz) return;
  int t = i / Bsz, b = i - t * Bsz;
  const f16* h = h2 + (long)i * 64;
  const float* w = Wt3 + t * 64;
  float s = bt3[t];
  for (int j = 0; j < 64; ++j) s += (float)h[j] * w[j];
  if (t < 2) s = 1.f / (1.f + __expf(-s));
  out[b * 3 + t] = s;
}

// ---------------------------------------------------------------------------
extern "C" void kernel_launch(void* const* d_in, const int* in_sizes, int n_in,
                              void* d_out, int out_size, void* d_ws, size_t ws_size,
                              hipStream_t stream) {
  const int*   x    = (const int*)  d_in[0];
  const float* emb  = (const float*)d_in[1];
  const float* We1a = (const float*)d_in[2];  const float* be1a = (const float*)d_in[3];
  const float* We2a = (const float*)d_in[4];  const float* be2a = (const float*)d_in[5];
  const float* We1b = (const float*)d_in[6];  const float* be1b = (const float*)d_in[7];
  const float* We2b = (const float*)d_in[8];  const float* be2b = (const float*)d_in[9];
  const float* Wg   = (const float*)d_in[10]; const float* bg   = (const float*)d_in[11];
  const float* Wf   = (const float*)d_in[12]; const float* bf   = (const float*)d_in[13];
  const float* Wq   = (const float*)d_in[14];
  const float* Wk   = (const float*)d_in[15];
  const float* Wv   = (const float*)d_in[16];
  const float* Wt1  = (const float*)d_in[17]; const float* bt1  = (const float*)d_in[18];
  const float* Wt2  = (const float*)d_in[19]; const float* bt2  = (const float*)d_in[20];
  const float* Wt3  = (const float*)d_in[21]; const float* bt3  = (const float*)d_in[22];
  float* out = (float*)d_out;

  // workspace bump allocator (deterministic layout every call)
  char* wsb = (char*)d_ws;
  size_t off = 0;
  auto alloc = [&](size_t bytes) -> char* {
    char* p = wsb + off;
    off = (off + bytes + 255) & ~(size_t)255;
    return p;
  };
  f16* we1a_t = (f16*)alloc(12L * 512 * 256 * 2);
  f16* we2a_t = (f16*)alloc(12L * 256 * 512 * 2);
  f16* we1b_t = (f16*)alloc(12L * 256 * 256 * 2);
  f16* we2b_t = (f16*)alloc(12L * 256 * 256 * 2);
  f16* wq_t   = (f16*)alloc(6L * 256 * 256 * 2);
  f16* wk_t   = (f16*)alloc(6L * 256 * 256 * 2);
  f16* wv_t   = (f16*)alloc(6L * 256 * 256 * 2);
  f16* wt1_t  = (f16*)alloc(3L * 128 * 256 * 2);
  f16* wt2_t  = (f16*)alloc(3L * 64 * 128 * 2);
  f16* Xf16   = (f16*)alloc((long)Tt * Bsz * Dd * 2);
  f16* h_a    = (f16*)alloc((long)Tt * 2 * Bsz * 512 * 2);
  f16* h_b    = (f16*)alloc((long)Tt * 2 * Bsz * 256 * 2);
  f16* e2     = (f16*)alloc((long)Tt * 4 * Bsz * 256 * 2);
  float* cgc  = (float*)alloc((long)Tt * Bsz * 256 * 4);
  f16* pair   = (f16*)alloc((long)Tt * Bsz * 2 * 256 * 2);
  f16* Qb     = (f16*)alloc((long)Tt * Bsz * 2 * 256 * 2);
  f16* Kb     = (f16*)alloc((long)Tt * Bsz * 2 * 256 * 2);
  f16* Vb     = (f16*)alloc((long)Tt * Bsz * 2 * 256 * 2);
  int* idx2   = (int*)alloc((long)Tt * Bsz * 4);
  f16* th1    = (f16*)alloc((long)Tt * Bsz * 128 * 2);
  f16* th2    = (f16*)alloc((long)Tt * Bsz * 64 * 2);
  (void)off; (void)ws_size; (void)in_sizes; (void)n_in; (void)out_size;

  auto convw = [&](const float* src, f16* dst, int nmat, int K, int N) {
    long total = (long)nmat * K * N;
    convw_kernel<<<dim3((unsigned)((total + 255) / 256)), 256, 0, stream>>>(src, dst, K, N, total);
  };
  auto gemm = [&](const f16* A, const f16* Wt, const float* bias, f16* outH, float* outF,
                  int M, int N, int K, int relu) {
    gemm_wmma<<<dim3(N / 64, M / 256), 256, 0, stream>>>(A, Wt, bias, outH, outF, M, N, K, relu);
  };

  // --- weight conversion (f32 [K][N] -> f16 [N][K]) ---
  convw(We1a, we1a_t, 12, 256, 512);
  convw(We2a, we2a_t, 12, 512, 256);
  convw(We1b, we1b_t, 12, 256, 256);
  convw(We2b, we2b_t, 12, 256, 256);
  convw(Wq,   wq_t,    6, 256, 256);
  convw(Wk,   wk_t,    6, 256, 256);
  convw(Wv,   wv_t,    6, 256, 256);
  convw(Wt1,  wt1_t,   3, 256, 128);
  convw(Wt2,  wt2_t,   3, 128,  64);

  // --- embedding ---
  embed_kernel<<<(Bsz * 16 + 255) / 256, 256, 0, stream>>>(x, emb, Xf16);

  // --- CGC levels ---
  for (int l = 0; l < 2; ++l) {
    for (int t = 0; t < Tt; ++t) {
      const f16* Xt = Xf16 + (long)t * Bsz * Dd;
      for (int e = 0; e < 2; ++e) {
        int m = (l * 3 + t) * 2 + e;
        // A experts: 256 -> 512 -> 256, relu both; second layer -> e2 slot 2e
        f16* ha = h_a + (long)(t * 2 + e) * Bsz * 512;
        gemm(Xt, we1a_t + (long)m * 512 * 256, be1a + m * 512, ha, nullptr, Bsz, 512, 256, 1);
        gemm(ha, we2a_t + (long)m * 256 * 512, be2a + m * 256,
             e2 + (long)(t * 4 + 2 * e) * Bsz * 256, nullptr, Bsz, 256, 512, 1);
        // B experts: 256 -> 256 -> 256; second layer -> e2 slot 2e+1
        f16* hb = h_b + (long)(t * 2 + e) * Bsz * 256;
        gemm(Xt, we1b_t + (long)m * 256 * 256, be1b + m * 256, hb, nullptr, Bsz, 256, 256, 1);
        gemm(hb, we2b_t + (long)m * 256 * 256, be2b + m * 256,
             e2 + (long)(t * 4 + 2 * e + 1) * Bsz * 256, nullptr, Bsz, 256, 256, 1);
      }
    }
    combine_kernel<<<Tt * Bsz / 8, 256, 0, stream>>>(
        Xf16, Wg + (long)l * Tt * Dd * 4, bg + l * Tt * 4,
        Wf + (long)l * Tt * Dd * 3, bf + l * Tt * 3, e2, cgc, pair, idx2);
    gather_kernel<<<(Tt * Bsz * Dd + 255) / 256, 256, 0, stream>>>(cgc, idx2, pair);
    for (int t = 0; t < Tt; ++t) {
      const f16* Pt = pair + (long)t * Bsz * 2 * Dd;
      long wo = (long)(l * 3 + t) * 256 * 256;
      gemm(Pt, wq_t + wo, nullptr, Qb + (long)t * Bsz * 2 * Dd, nullptr, 2 * Bsz, 256, 256, 0);
      gemm(Pt, wk_t + wo, nullptr, Kb + (long)t * Bsz * 2 * Dd, nullptr, 2 * Bsz, 256, 256, 0);
      gemm(Pt, wv_t + wo, nullptr, Vb + (long)t * Bsz * 2 * Dd, nullptr, 2 * Bsz, 256, 256, 0);
    }
    attn_kernel<<<Tt * Bsz / 8, 256, 0, stream>>>(Qb, Kb, Vb, Xf16);  // writes next-level X
  }

  // --- towers ---
  for (int t = 0; t < Tt; ++t) {
    gemm(Xf16 + (long)t * Bsz * Dd, wt1_t + (long)t * 128 * 256, bt1 + t * 128,
         th1 + (long)t * Bsz * 128, nullptr, Bsz, 128, 256, 1);
    gemm(th1 + (long)t * Bsz * 128, wt2_t + (long)t * 64 * 128, bt2 + t * 64,
         th2 + (long)t * Bsz * 64, nullptr, Bsz, 64, 128, 1);
  }
  tower_out_kernel<<<(Tt * Bsz + 255) / 256, 256, 0, stream>>>(th2, Wt3, bt3, out);
}
